// Model_5506148074015
// MI455X (gfx1250) — matrix operand order repaired and checked by reading, not compiled, not run on hardware
//
#include <hip/hip_runtime.h>
#include <hip/hip_bf16.h>
#include <math.h>

// ---------------------------------------------------------------------------
// Types
// ---------------------------------------------------------------------------
typedef __attribute__((ext_vector_type(16))) __bf16 v16bf;
typedef __attribute__((ext_vector_type(8)))  __bf16 v8bf;
typedef __attribute__((ext_vector_type(8)))  float  v8f;

struct c32 { float x, y; };
__device__ __forceinline__ c32 mkc(float a, float b) { c32 r; r.x = a; r.y = b; return r; }
__device__ __forceinline__ c32 cadd(c32 a, c32 b) { return mkc(a.x + b.x, a.y + b.y); }
__device__ __forceinline__ c32 csub(c32 a, c32 b) { return mkc(a.x - b.x, a.y - b.y); }
__device__ __forceinline__ c32 cmul(c32 a, c32 b) {
  return mkc(a.x * b.x - a.y * b.y, a.x * b.y + a.y * b.x);
}
__device__ __forceinline__ c32 cscale(c32 a, float s) { return mkc(a.x * s, a.y * s); }
__device__ __forceinline__ c32 cinv(c32 a) {
  float d = 1.0f / (a.x * a.x + a.y * a.y);
  return mkc(a.x * d, -a.y * d);
}

// ---------------------------------------------------------------------------
// Stockham radix-2 FFT in LDS (self-sorting, ping-pong buffers).
// sign = -1: forward DFT, sign = +1: inverse DFT (unscaled).
// Returns pointer to the buffer holding the result. Barrier after each stage.
// ---------------------------------------------------------------------------
template <int NTHR>
__device__ c32* fft_stockham(c32* x, c32* y, int n, float sign) {
  const int tid = threadIdx.x;
  int ncur = n;
  int s = 1, ls = 0;
  while (ncur > 1) {
    const int m = ncur >> 1;
    const float theta = sign * 6.28318530717958647692f / (float)ncur;
    for (int t = tid; t < (n >> 1); t += NTHR) {
      const int p = t >> ls;
      const int q = t & (s - 1);
      c32 a = x[q + s * p];
      c32 b = x[q + s * (p + m)];
      float wi, wr;
      __sincosf(theta * (float)p, &wi, &wr);
      c32 d = csub(a, b);
      y[q + s * (2 * p)]     = cadd(a, b);
      y[q + s * (2 * p + 1)] = mkc(d.x * wr - d.y * wi, d.x * wi + d.y * wr);
    }
    __syncthreads();
    c32* tmp = x; x = y; y = tmp;
    ncur = m; s <<= 1; ++ls;
  }
  return x;
}

// ---------------------------------------------------------------------------
// WMMA fragment loaders (CDNA5 16x16x32 bf16 layouts, ISA 7.12.2)
// A: row-major activations (M x K), lane(r=l&15, g=l>>4) holds row m0+r,
//    k = g*16 + {0..15} (same contiguous order as B).
// B: row-major weight W[N][K] (computing A @ W^T), lane(c=l&15, g=l>>4)
//    holds col n0+c, k = g*16 + {0..15} (one contiguous 32B run).
// ---------------------------------------------------------------------------
__device__ __forceinline__ v16bf load_a_frag(const __bf16* __restrict__ A, int lda,
                                             int m0, int k0, int lane) {
  const int r = lane & 15, g = lane >> 4;
  const __bf16* base = A + (size_t)(m0 + r) * lda + k0 + g * 16;
  v8bf lo = *(const v8bf*)(base);
  v8bf hi = *(const v8bf*)(base + 8);
  v16bf out;
#pragma unroll
  for (int i = 0; i < 8; ++i) { out[i] = lo[i]; out[i + 8] = hi[i]; }
  return out;
}

__device__ __forceinline__ v16bf load_b_frag(const __bf16* __restrict__ W, int ldw,
                                             int n0, int k0, int lane) {
  const int c = lane & 15, g = lane >> 4;
  return *(const v16bf*)(W + (size_t)(n0 + c) * ldw + k0 + g * 16);
}

// ---------------------------------------------------------------------------
// f32 -> bf16 conversion
// ---------------------------------------------------------------------------
__global__ void __launch_bounds__(256)
f32_to_bf16_kernel(const float* __restrict__ s, __bf16* __restrict__ d, int n) {
  int i = blockIdx.x * 256 + threadIdx.x;
  if (i < n) d[i] = (__bf16)s[i];
}

// ---------------------------------------------------------------------------
// C = A(bf16, MxK) @ W(bf16, NxK)^T + bias  -> C(f32, MxN)
// 8 waves/block, one 16x16 tile per wave. Tile counts divisible by 8.
// ---------------------------------------------------------------------------
__global__ void __launch_bounds__(256)
gemm_bias_wmma(const __bf16* __restrict__ A, int lda,
               const __bf16* __restrict__ W, int ldw,
               const float* __restrict__ bias,
               float* __restrict__ C, int ldc,
               int ntiles_n, int K) {
  const int lane = threadIdx.x & 31;
  const int wave = threadIdx.x >> 5;
  const int tile = blockIdx.x * 8 + wave;
  const int nt = tile % ntiles_n;
  const int mt = tile / ntiles_n;
  const int m0 = mt * 16, n0 = nt * 16;

  v8f acc = {};
  for (int k0 = 0; k0 < K; k0 += 32) {
    if (k0 + 32 < K)
      __builtin_prefetch(A + (size_t)(m0 + (lane & 15)) * lda + k0 + 32, 0, 3);
    v16bf a = load_a_frag(A, lda, m0, k0, lane);
    v16bf b = load_b_frag(W, ldw, n0, k0, lane);
    acc = __builtin_amdgcn_wmma_f32_16x16x32_bf16(false, a, false, b,
                                                  (short)0, acc, false, false);
  }
  const int c = lane & 15, g = lane >> 4;
  const float bn = bias[n0 + c];
#pragma unroll
  for (int j = 0; j < 8; ++j)
    C[(size_t)(m0 + g * 8 + j) * ldc + (n0 + c)] = acc[j] + bn;
}

// ---------------------------------------------------------------------------
// Gated MLP + residual (in place on H):
//   H += (A@W1^T + b1) * sigmoid(A@W2^T + b2),  all dims 128, M=16384
// One A-fragment feeds two WMMAs per K-step (8 WMMA per tile, unrolled).
// ---------------------------------------------------------------------------
__global__ void __launch_bounds__(256)
mlp_gated_wmma(const __bf16* __restrict__ A,
               const __bf16* __restrict__ W1, const __bf16* __restrict__ W2,
               const float* __restrict__ b1v, const float* __restrict__ b2v,
               float* __restrict__ H) {
  const int lane = threadIdx.x & 31;
  const int wave = threadIdx.x >> 5;
  const int tile = blockIdx.x * 8 + wave;
  const int nt = tile & 7;         // 128/16 = 8 n-tiles
  const int mt = tile >> 3;
  const int m0 = mt * 16, n0 = nt * 16;

  v8f acc1 = {}, acc2 = {};
#pragma unroll
  for (int kk = 0; kk < 4; ++kk) {
    const int k0 = kk * 32;
    v16bf a  = load_a_frag(A, 128, m0, k0, lane);
    v16bf f1 = load_b_frag(W1, 128, n0, k0, lane);
    v16bf f2 = load_b_frag(W2, 128, n0, k0, lane);
    acc1 = __builtin_amdgcn_wmma_f32_16x16x32_bf16(false, a, false, f1,
                                                   (short)0, acc1, false, false);
    acc2 = __builtin_amdgcn_wmma_f32_16x16x32_bf16(false, a, false, f2,
                                                   (short)0, acc2, false, false);
  }
  const int c = lane & 15, g = lane >> 4;
  const float bb1 = b1v[n0 + c], bb2 = b2v[n0 + c];
#pragma unroll
  for (int j = 0; j < 8; ++j) {
    const size_t idx = (size_t)(m0 + g * 8 + j) * 128 + (n0 + c);
    const float g1 = acc1[j] + bb1;
    const float g2 = acc2[j] + bb2;
    H[idx] = H[idx] + g1 * (1.0f / (1.0f + __expf(-g2)));
  }
}

// ---------------------------------------------------------------------------
// LayerNorm over H=128 per token; 2 tokens per 256-thread block.
// ---------------------------------------------------------------------------
__global__ void __launch_bounds__(256)
layernorm_kernel(const float* __restrict__ Hin, const float* __restrict__ nw,
                 const float* __restrict__ nb, float* __restrict__ Hn) {
  __shared__ float red[256];
  const int sub = threadIdx.x >> 7;
  const int ch  = threadIdx.x & 127;
  const size_t token = (size_t)blockIdx.x * 2 + sub;
  const float v = Hin[token * 128 + ch];

  red[threadIdx.x] = v;
  __syncthreads();
#pragma unroll
  for (int off = 64; off >= 1; off >>= 1) {
    if (ch < off) red[threadIdx.x] += red[threadIdx.x + off];
    __syncthreads();
  }
  const float mu = red[sub * 128] * (1.0f / 128.0f);
  __syncthreads();
  const float dv = v - mu;
  red[threadIdx.x] = dv * dv;
  __syncthreads();
#pragma unroll
  for (int off = 64; off >= 1; off >>= 1) {
    if (ch < off) red[threadIdx.x] += red[threadIdx.x + off];
    __syncthreads();
  }
  const float var = red[sub * 128] * (1.0f / 128.0f);
  Hn[token * 128 + ch] = dv * rsqrtf(var + 1e-5f) * nw[ch] + nb[ch];
}

// ---------------------------------------------------------------------------
// S4 kernel generation, one workgroup per h channel:
//   Cauchy sums (N=64) + Woodbury at the 2048 roots of unity  -> at_roots
//   IFFT_2048 -> real K -> zero-pad -> FFT_4096 -> Kf[h]
// Dynamic LDS: 2 x 4096 c32 ping-pong + 5 x 64 c32 parameter arrays.
// ---------------------------------------------------------------------------
__global__ void __launch_bounds__(256)
s4_kernel_gen(const float* __restrict__ lr, const float* __restrict__ li,
              const float* __restrict__ pr, const float* __restrict__ pi,
              const float* __restrict__ br, const float* __restrict__ bi,
              const float* __restrict__ cr, const float* __restrict__ ci,
              const float* __restrict__ logstep, c32* __restrict__ Kf) {
  extern __shared__ unsigned char smem_raw[];
  c32* bufA = (c32*)smem_raw;     // 4096
  c32* bufB = bufA + 4096;        // 4096
  c32* lam  = bufB + 4096;        // 64
  c32* v00  = lam + 64;
  c32* v01  = v00 + 64;
  c32* v10  = v01 + 64;
  c32* v11  = v10 + 64;

  const int h = blockIdx.x;
  const int tid = threadIdx.x;
  const int L = 2048;

  if (tid < 64) {
    const int idx = h * 64 + tid;
    const float LR = lr[idx], LI = li[idx];
    const float PR = pr[idx], PI = pi[idx];
    const float BR = br[idx], BI = bi[idx];
    const float CR = cr[idx], CI = ci[idx];
    lam[tid] = mkc(LR, LI);
    v00[tid] = mkc(CR * BR + CI * BI, CR * BI - CI * BR);  // conj(C)*B
    v01[tid] = mkc(CR * PR + CI * PI, CR * PI - CI * PR);  // conj(C)*P
    v10[tid] = mkc(PR * BR + PI * BI, PR * BI - PI * BR);  // conj(P)*B
    v11[tid] = mkc(PR * PR + PI * PI, 0.0f);               // conj(P)*P
  }
  const float two_over_step = 2.0f * __expf(-logstep[h]);
  __syncthreads();

  for (int l = tid; l < L; l += 256) {
    float wi, wr;
    __sincosf(-6.283185307179586f * (float)l / (float)L, &wi, &wr);
    const c32 onep  = mkc(1.0f + wr, wi);
    const c32 ionep = cinv(onep);
    const c32 g  = cscale(cmul(mkc(1.0f - wr, -wi), ionep), two_over_step);
    const c32 c2 = cscale(ionep, 2.0f);
    c32 k00 = mkc(0, 0), k01 = mkc(0, 0), k10 = mkc(0, 0), k11 = mkc(0, 0);
#pragma unroll 4
    for (int n = 0; n < 64; ++n) {
      const c32 r = cinv(mkc(g.x - lam[n].x, g.y - lam[n].y));
      k00 = cadd(k00, cmul(v00[n], r));
      k01 = cadd(k01, cmul(v01[n], r));
      k10 = cadd(k10, cmul(v10[n], r));
      k11 = cadd(k11, cmul(v11[n], r));
    }
    const c32 corr = cmul(cmul(k01, k10), cinv(mkc(1.0f + k11.x, k11.y)));
    bufA[l] = cmul(c2, csub(k00, corr));
  }
  __syncthreads();

  c32* r = fft_stockham<256>(bufA, bufB, 2048, +1.0f);   // ifft (unscaled)
  c32* dst = (r == bufA) ? bufB : bufA;
  const float inv = 1.0f / 2048.0f;
  for (int l = tid; l < 4096; l += 256)
    dst[l] = (l < 2048) ? mkc(r[l].x * inv, 0.0f) : mkc(0.0f, 0.0f);
  __syncthreads();

  c32* r2 = fft_stockham<256>(dst, r, 4096, -1.0f);      // forward 4096
  for (int l = tid; l < 4096; l += 256) Kf[(size_t)h * 4096 + l] = r2[l];
}

// ---------------------------------------------------------------------------
// Causal conv via FFT + D skip + GELU, fused; one workgroup per (b,h).
// Reads hn f32 (B,L,H), writes gelu(y) bf16 (B,L,H) as next-GEMM A operand.
// Dynamic LDS: 2 x 4096 c32 + 2048 f32 copy of u.
// ---------------------------------------------------------------------------
__global__ void __launch_bounds__(256)
s4_conv_gelu(const float* __restrict__ Hn, const c32* __restrict__ Kf,
             const float* __restrict__ dvec, __bf16* __restrict__ Y) {
  extern __shared__ unsigned char smem_raw[];
  c32* bufA = (c32*)smem_raw;
  c32* bufB = bufA + 4096;
  float* us = (float*)(bufB + 4096);

  const int b = blockIdx.x >> 7;
  const int h = blockIdx.x & 127;
  const int tid = threadIdx.x;
  const int L = 2048, Hc = 128;

  const float* up = Hn + (size_t)b * L * Hc + h;
  for (int l = tid; l < 4096; l += 256) {
    if (l < L) { const float u = up[(size_t)l * Hc]; us[l] = u; bufA[l] = mkc(u, 0.0f); }
    else       bufA[l] = mkc(0.0f, 0.0f);
  }
  __syncthreads();

  c32* r = fft_stockham<256>(bufA, bufB, 4096, -1.0f);
  const c32* kf = Kf + (size_t)h * 4096;
  for (int l = tid; l < 4096; l += 256) r[l] = cmul(r[l], kf[l]);
  __syncthreads();

  c32* o = fft_stockham<256>(r, (r == bufA) ? bufB : bufA, 4096, +1.0f);
  const float dh = dvec[h];
  const float inv = 1.0f / 4096.0f;
  for (int l = tid; l < L; l += 256) {
    const float v = o[l].x * inv + us[l] * dh;
    const float ge = 0.5f * v * (1.0f + erff(v * 0.70710678118654752f));
    Y[((size_t)b * L + l) * Hc + h] = (__bf16)ge;
  }
}

// ---------------------------------------------------------------------------
// Host launch
// ---------------------------------------------------------------------------
extern "C" void kernel_launch(void* const* d_in, const int* in_sizes, int n_in,
                              void* d_out, int out_size, void* d_ws, size_t ws_size,
                              hipStream_t stream) {
  (void)in_sizes; (void)n_in; (void)out_size; (void)ws_size;
  const float* x        = (const float*)d_in[0];
  const float* enc_w    = (const float*)d_in[2];
  const float* enc_b    = (const float*)d_in[3];
  const float* dec_w    = (const float*)d_in[4];
  const float* dec_b    = (const float*)d_in[5];
  const float* lam_re   = (const float*)d_in[6];
  const float* lam_im   = (const float*)d_in[7];
  const float* p_re     = (const float*)d_in[8];
  const float* p_im     = (const float*)d_in[9];
  const float* b_re     = (const float*)d_in[10];
  const float* b_im     = (const float*)d_in[11];
  const float* c_re     = (const float*)d_in[12];
  const float* c_im     = (const float*)d_in[13];
  const float* dvec     = (const float*)d_in[14];
  const float* log_step = (const float*)d_in[15];
  const float* norm_w   = (const float*)d_in[16];
  const float* norm_b   = (const float*)d_in[17];
  const float* w1       = (const float*)d_in[18];
  const float* b1       = (const float*)d_in[19];
  const float* w2       = (const float*)d_in[20];
  const float* b2       = (const float*)d_in[21];
  float* out = (float*)d_out;

  const int B = 8, L = 2048, Hc = 128, IN = 64, OUT = 64, NL = 4;
  const int M = B * L;                          // 16384 tokens

  // Workspace layout (≈ 24.3 MB, 256B-aligned sections)
  char* ws = (char*)d_ws;
  float*  h    = (float*)(ws);                         // 8 MB
  float*  hn   = (float*)(ws + (size_t)(8u << 20));    // 8 MB
  __bf16* abf  = (__bf16*)(ws + (size_t)(16u << 20));  // 4 MB bf16 activations
  c32*    Kf   = (c32*)(ws + (size_t)(20u << 20));     // 4 MB spectra
  __bf16* encw = (__bf16*)(ws + (size_t)(24u << 20));
  __bf16* decw = encw + Hc * IN;
  __bf16* w1b  = decw + OUT * Hc;
  __bf16* w2b  = w1b + (size_t)NL * Hc * Hc;

  const size_t smem_gen  = (size_t)(8192 + 5 * 64) * sizeof(c32);      // ~66.5 KB
  const size_t smem_conv = (size_t)8192 * sizeof(c32) + 2048 * 4;      //  72 KB

  // Weight + input conversions to bf16
  f32_to_bf16_kernel<<<(Hc * IN + 255) / 256, 256, 0, stream>>>(enc_w, encw, Hc * IN);
  f32_to_bf16_kernel<<<(OUT * Hc + 255) / 256, 256, 0, stream>>>(dec_w, decw, OUT * Hc);
  f32_to_bf16_kernel<<<(NL * Hc * Hc + 255) / 256, 256, 0, stream>>>(w1, w1b, NL * Hc * Hc);
  f32_to_bf16_kernel<<<(NL * Hc * Hc + 255) / 256, 256, 0, stream>>>(w2, w2b, NL * Hc * Hc);
  f32_to_bf16_kernel<<<(M * IN + 255) / 256, 256, 0, stream>>>(x, abf, M * IN);

  // Encoder: h = x @ enc_w^T + enc_b   (tiles: 1024 x 8 -> 1024 blocks)
  gemm_bias_wmma<<<(M / 16) * (Hc / 16) / 8, 256, 0, stream>>>(
      abf, IN, encw, IN, enc_b, h, Hc, Hc / 16, IN);

  for (int i = 0; i < NL; ++i) {
    layernorm_kernel<<<M / 2, 256, 0, stream>>>(h, norm_w + i * Hc, norm_b + i * Hc, hn);

    s4_kernel_gen<<<Hc, 256, smem_gen, stream>>>(
        lam_re + (size_t)i * Hc * 64, lam_im + (size_t)i * Hc * 64,
        p_re + (size_t)i * Hc * 64,   p_im + (size_t)i * Hc * 64,
        b_re + (size_t)i * Hc * 64,   b_im + (size_t)i * Hc * 64,
        c_re + (size_t)i * Hc * 64,   c_im + (size_t)i * Hc * 64,
        log_step + i * Hc, Kf);

    s4_conv_gelu<<<B * Hc, 256, smem_conv, stream>>>(hn, Kf, dvec + i * Hc, abf);

    mlp_gated_wmma<<<(M / 16) * (Hc / 16) / 8, 256, 0, stream>>>(
        abf, w1b + (size_t)i * Hc * Hc, w2b + (size_t)i * Hc * Hc,
        b1 + i * Hc, b2 + i * Hc, h);
  }

  // Decoder: out = h @ dec_w^T + dec_b   (tiles: 1024 x 4 -> 512 blocks)
  f32_to_bf16_kernel<<<(M * Hc + 255) / 256, 256, 0, stream>>>(h, abf, M * Hc);
  gemm_bias_wmma<<<(M / 16) * (OUT / 16) / 8, 256, 0, stream>>>(
      abf, Hc, decw, Hc, dec_b, out, OUT, OUT / 16, Hc);
}
